// Net_4449586119027
// MI455X (gfx1250) — matrix operand order repaired and checked by reading, not compiled, not run on hardware
//
#include <hip/hip_runtime.h>
#include <hip/hip_bf16.h>
#include <math.h>

#define N_RES      30000
#define N_DOM      6
#define BATCH      128
#define HID        350
#define NB_PER_RES 10
#define HID_PAD    352   // HID padded to multiple of 32 for WMMA K-loop

typedef __attribute__((ext_vector_type(16))) _Float16 v16h;
typedef __attribute__((ext_vector_type(8)))  float    v8f;

// ---------------------------------------------------------------------------
// Kernel 1: per-residue window product + softmax over domains -> attn (N_RES,6)
// ---------------------------------------------------------------------------
__global__ void attn_kernel(const float* __restrict__ weights,   // (3009, 6)
                            const int*   __restrict__ bs,        // (N_RES, 10)
                            float*       __restrict__ attn) {    // (N_RES, 6)
  int r = blockIdx.x * blockDim.x + threadIdx.x;
  if (r >= N_RES) return;
  const int* bsr = bs + r * NB_PER_RES;
  float prod[N_DOM];
#pragma unroll
  for (int d = 0; d < N_DOM; ++d) prod[d] = 1.0f;
#pragma unroll
  for (int w = 0; w < NB_PER_RES; ++w) {
    int base = bsr[w] * N_DOM;
#pragma unroll
    for (int d = 0; d < N_DOM; ++d) prod[d] *= weights[base + d];
  }
  float m = prod[0];
#pragma unroll
  for (int d = 1; d < N_DOM; ++d) m = fmaxf(m, prod[d]);
  float e[N_DOM];
  float s = 0.0f;
#pragma unroll
  for (int d = 0; d < N_DOM; ++d) { e[d] = expf(prod[d] - m); s += e[d]; }
  float inv = 1.0f / s;
#pragma unroll
  for (int d = 0; d < N_DOM; ++d) attn[r * N_DOM + d] = e[d] * inv;
}

// ---------------------------------------------------------------------------
// Kernel 2: GEMM1 + bias + relu via v_wmma_f32_16x16x32_f16
//   H(128,352 f16, zero-padded) = relu(L(128,18) @ W1(18,350) + b1)
// One wave per 16x16 output tile; 8 M-tiles x 22 N-tiles = 176 waves.
// ---------------------------------------------------------------------------
__global__ void mlp1_wmma(const float* __restrict__ L,
                          const float* __restrict__ W1,
                          const float* __restrict__ b1,
                          _Float16*    __restrict__ H) {
  int tile = blockIdx.x;              // 0..175
  int mt = tile / 22, nt = tile % 22;
  int m0 = mt * 16, n0 = nt * 16;
  int lane = threadIdx.x;
  int half = lane >> 4;
  int l15  = lane & 15;
  int m = m0 + l15;                   // A-matrix row for this lane
  int n = n0 + l15;                   // B/C/D column for this lane

  v16h a, bm;
  // A tile: 16x32 f16, K padded 18 -> 32 with zeros
#pragma unroll
  for (int v = 0; v < 8; ++v) {
#pragma unroll
    for (int p = 0; p < 2; ++p) {
      int kl = (v < 4) ? (2 * v + p + 8 * half)
                       : (16 + 2 * (v - 4) + p + 8 * half);
      float x = (kl < 18) ? L[m * 18 + kl] : 0.0f;
      a[2 * v + p] = (_Float16)x;
    }
  }
  // B tile: 32x16 f16 from W1 (18x350), zero-pad K>=18 and N>=350
#pragma unroll
  for (int s = 0; s < 16; ++s) {
    int k = (s < 8) ? (s + 8 * half) : (8 + s + 8 * half);
    float x = (k < 18 && n < HID) ? W1[k * HID + n] : 0.0f;
    bm[s] = (_Float16)x;
  }

  v8f c = {};
  c = __builtin_amdgcn_wmma_f32_16x16x32_f16(false, a, false, bm,
                                             (short)0, c, false, false);

  float bias = (n < HID) ? b1[n] : 0.0f;
#pragma unroll
  for (int i = 0; i < 8; ++i) {
    int M = m0 + i + 8 * half;
    float v_ = fmaxf(c[i] + bias, 0.0f);
    // write padded columns as exact zeros so GEMM2 K-loop needs no edge logic
    H[M * HID_PAD + n] = (n < HID) ? (_Float16)v_ : (_Float16)0.0f;
  }
}

// ---------------------------------------------------------------------------
// Kernel 3: GEMM2 + bias via v_wmma_f32_16x16x32_f16
//   S(128,18 f32) = H(128,352 f16) @ W2(350,18) + b2
// 8 M-tiles x 2 N-tiles = 16 waves; 11 K-steps of 32 each.
// ---------------------------------------------------------------------------
__global__ void mlp2_wmma(const _Float16* __restrict__ H,
                          const float*    __restrict__ W2,
                          const float*    __restrict__ b2,
                          float*          __restrict__ S) {
  int tile = blockIdx.x;              // 0..15
  int mt = tile >> 1, nt = tile & 1;
  int m0 = mt * 16, n0 = nt * 16;
  int lane = threadIdx.x;
  int half = lane >> 4;
  int l15  = lane & 15;
  int m = m0 + l15;
  int n = n0 + l15;

  v8f c = {};
  for (int kt = 0; kt < HID_PAD / 32; ++kt) {
    v16h a, bm;
#pragma unroll
    for (int v = 0; v < 8; ++v) {
#pragma unroll
      for (int p = 0; p < 2; ++p) {
        int kl = (v < 4) ? (2 * v + p + 8 * half)
                         : (16 + 2 * (v - 4) + p + 8 * half);
        a[2 * v + p] = H[m * HID_PAD + kt * 32 + kl];
      }
    }
#pragma unroll
    for (int s = 0; s < 16; ++s) {
      int k = kt * 32 + ((s < 8) ? (s + 8 * half) : (8 + s + 8 * half));
      float x = (k < HID && n < 18) ? W2[k * 18 + n] : 0.0f;
      bm[s] = (_Float16)x;
    }
    c = __builtin_amdgcn_wmma_f32_16x16x32_f16(false, a, false, bm,
                                               (short)0, c, false, false);
  }

  if (n < 18) {
    float bias = b2[n];
#pragma unroll
    for (int i = 0; i < 8; ++i) {
      int M = m0 + i + 8 * half;
      S[M * 18 + n] = c[i] + bias;
    }
  }
}

// ---------------------------------------------------------------------------
// Kernel 4: translation_vectors = scalars(B,6,3) @ local_frame(3,3)
// ---------------------------------------------------------------------------
__global__ void tv_kernel(const float* __restrict__ S,    // (B, 18)
                          const float* __restrict__ LF,   // (3, 3)
                          float*       __restrict__ TV) { // (B, 18)
  int idx = blockIdx.x * blockDim.x + threadIdx.x;        // b*6 + d
  if (idx >= BATCH * N_DOM) return;
  int b = idx / N_DOM, d = idx % N_DOM;
  const float* s = S + b * 18 + d * 3;
  float* t = TV + b * 18 + d * 3;
#pragma unroll
  for (int cc = 0; cc < 3; ++cc) {
    t[cc] = s[0] * LF[0 * 3 + cc] + s[1] * LF[1 * 3 + cc] + s[2] * LF[2 * 3 + cc];
  }
}

// ---------------------------------------------------------------------------
// Kernel 5 (bandwidth-dominant): per (b, residue)
//   tpr[b][r][:]  = sum_d attn[r][d] * tv[b][d][:]
//   ns[b][3r+j][:] = atoms[3r+j][:] + tpr[b][r][:]   j=0..2
// Wave writes 9 contiguous dwords/lane -> fully coalesced 1152B bursts.
// ---------------------------------------------------------------------------
__global__ void deform_kernel(const float* __restrict__ attn,   // (N_RES, 6)
                              const float* __restrict__ TV,     // (B, 18)
                              const float* __restrict__ atoms,  // (3*N_RES, 3)
                              float*       __restrict__ ns,     // (B, 3*N_RES, 3)
                              float*       __restrict__ tpr) {  // (B, N_RES, 3)
  int r = blockIdx.x * blockDim.x + threadIdx.x;
  if (r >= N_RES) return;
  int b = blockIdx.y;
  const float* tvb = TV + b * 18;     // wave-uniform -> scalar loads
  float t0 = 0.f, t1 = 0.f, t2 = 0.f;
#pragma unroll
  for (int d = 0; d < N_DOM; ++d) {
    float w = attn[r * N_DOM + d];
    t0 += w * tvb[3 * d + 0];
    t1 += w * tvb[3 * d + 1];
    t2 += w * tvb[3 * d + 2];
  }
  size_t tbase = ((size_t)b * N_RES + r) * 3;
  tpr[tbase + 0] = t0;
  tpr[tbase + 1] = t1;
  tpr[tbase + 2] = t2;
  size_t abase = (size_t)r * 9;
  size_t obase = (size_t)b * (3u * N_RES * 3u) + abase;
#pragma unroll
  for (int j = 0; j < 3; ++j) {
    ns[obase + 3 * j + 0] = atoms[abase + 3 * j + 0] + t0;
    ns[obase + 3 * j + 1] = atoms[abase + 3 * j + 1] + t1;
    ns[obase + 3 * j + 2] = atoms[abase + 3 * j + 2] + t2;
  }
}

// ---------------------------------------------------------------------------
extern "C" void kernel_launch(void* const* d_in, const int* in_sizes, int n_in,
                              void* d_out, int out_size, void* d_ws, size_t ws_size,
                              hipStream_t stream) {
  const float* latent  = (const float*)d_in[0];   // (128, 18)
  const float* weights = (const float*)d_in[1];   // (3009, 6)
  const float* atoms   = (const float*)d_in[2];   // (90000, 3)
  const float* LF      = (const float*)d_in[3];   // (3, 3)
  const float* w1      = (const float*)d_in[4];   // (18, 350)
  const float* b1      = (const float*)d_in[5];   // (350,)
  const float* w2      = (const float*)d_in[6];   // (350, 18)
  const float* b2      = (const float*)d_in[7];   // (18,)
  const int*   bs      = (const int*)d_in[8];     // (30000, 10)

  float* out  = (float*)d_out;
  float* ns   = out;                                        // 128*90000*3
  float* attn = out + (size_t)BATCH * 3 * N_RES * 3;        // 30000*6
  float* tpr  = attn + (size_t)N_RES * N_DOM;               // 128*30000*3

  char* ws = (char*)d_ws;
  _Float16* H  = (_Float16*)ws;                             // 128*352*2  = 90112 B
  float*    S  = (float*)(ws + (size_t)BATCH * HID_PAD * 2);        // 9216 B
  float*    TV = (float*)(ws + (size_t)BATCH * HID_PAD * 2 + 9216); // 9216 B

  attn_kernel<<<(N_RES + 255) / 256, 256, 0, stream>>>(weights, bs, attn);
  mlp1_wmma<<<8 * 22, 32, 0, stream>>>(latent, w1, b1, H);
  mlp2_wmma<<<8 * 2, 32, 0, stream>>>(H, w2, b2, S);
  tv_kernel<<<(BATCH * N_DOM + 255) / 256, 256, 0, stream>>>(S, LF, TV);
  deform_kernel<<<dim3((N_RES + 255) / 256, BATCH), 256, 0, stream>>>(
      attn, TV, atoms, ns, tpr);
}
